// GraphPropLayer_72730976190873
// MI455X (gfx1250) — compile-verified, matrix operand order checked by reading, not run on hardware
//
#include <hip/hip_runtime.h>

typedef __attribute__((ext_vector_type(16))) __bf16 bf16x16;
typedef __attribute__((ext_vector_type(8)))  float  v8f;

union FragU { uint4 q[2]; bf16x16 v; };

__device__ __forceinline__ unsigned short f2bf(float f) {
  unsigned int u = __float_as_uint(f);
  u += 0x7FFFu + ((u >> 16) & 1u);          // round-to-nearest-even
  return (unsigned short)(u >> 16);
}
__device__ __forceinline__ float bf2f(unsigned short s) {
  return __uint_as_float(((unsigned int)s) << 16);
}

// Fragment load (A or B operand), 16-bit 16x32 layout:
// lane<16 -> K {0..7,16..23}, lane>=16 -> K {8..15,24..31}
// => two 16B loads at +0 and +32 bytes from the lane's row/col base.
__device__ __forceinline__ bf16x16 load_frag_g(const unsigned short* p) {
  FragU f;
  f.q[0] = *(const uint4*)(p);
  f.q[1] = *(const uint4*)(p + 16);   // +16 elements = +32 bytes
  return f.v;
}
__device__ __forceinline__ bf16x16 load_frag_lds(const char* p) {
  FragU f;
  f.q[0] = *(const uint4*)(p);
  f.q[1] = *(const uint4*)(p + 32);
  return f.v;
}

#define WMMA_BF16(a, b, c) \
  __builtin_amdgcn_wmma_f32_16x16x32_bf16(false, (a), false, (b), (short)0, (c), false, false)

#define HSTRIDE 272   // 128 bf16 + 16B pad -> conflict-free ds_load_b128

// ---------------------------------------------------------------------------
__global__ __launch_bounds__(256) void zero_f32_kernel(float4* p, int n4) {
  int i = blockIdx.x * 256 + threadIdx.x;
  if (i < n4) p[i] = (float4){0.f, 0.f, 0.f, 0.f};
}

// dst[n*K + k] = bf16(src[k*Nout + n])   (src row-major [K][Nout])
__global__ __launch_bounds__(256) void transpose_bf16_kernel(
    const float* __restrict__ src, unsigned short* __restrict__ dst, int K, int Nout) {
  int i = blockIdx.x * 256 + threadIdx.x;
  if (i >= K * Nout) return;
  int n = i / K, k = i % K;
  dst[n * K + k] = f2bf(src[k * Nout + n]);
}

// ---------------------------------------------------------------------------
// Edge message MLP (both directions) + atomic scatter into agg.
// Block: 256 threads = 8 waves; 64 edges -> 128 message rows.
// Waves 0-3: dir1 ([a|b|ef] -> agg[to]);  waves 4-7: dir2 ([b|a|ef] -> agg[from]).
// Both directions share the same three staged 64x128 bf16 input blocks.
__global__ __launch_bounds__(256) void edge_msg_kernel(
    const float* __restrict__ ns, const float* __restrict__ ef,
    const int* __restrict__ fi, const int* __restrict__ ti,
    const unsigned short* __restrict__ w1t, const float* __restrict__ bias1,
    const unsigned short* __restrict__ w2t, const float* __restrict__ bias2,
    const unsigned short* __restrict__ w3t, const float* __restrict__ bias3,
    float* __restrict__ agg, int E) {
  extern __shared__ char smem[];
  char* Ab = smem;                  // 64 x HSTRIDE : ns[fi[e]]
  char* Bb = Ab + 64 * HSTRIDE;     // 64 x HSTRIDE : ns[ti[e]]
  char* Fb = Bb + 64 * HSTRIDE;     // 64 x HSTRIDE : ef[e]
  char* Hs = Fb + 64 * HSTRIDE;     // 128 x HSTRIDE: H1, then overwritten by H2 (per-wave
                                    // strips; DS ops within a wave are in-order -> safe)

  const int tid = threadIdx.x;
  const int tileBase = blockIdx.x * 64;

  // ---- stage the three shared input blocks as bf16 ----
  for (int q = tid; q < 3 * 64 * 32; q += 256) {
    int arr = q >> 11;            // 0:A 1:B 2:EF  (2048 quads each)
    int rem = q & 2047;
    int row = rem >> 5;
    int c4  = (rem & 31) * 4;
    int e   = tileBase + row;
    int ec  = (e < E) ? e : 0;
    const float* src;
    char* dst;
    if (arr == 0)      { src = ns + fi[ec] * 128; dst = Ab; }
    else if (arr == 1) { src = ns + ti[ec] * 128; dst = Bb; }
    else               { src = ef + ec * 128;     dst = Fb; }
    float4 v = *(const float4*)(src + c4);
    uint2 pk;
    pk.x = (unsigned)f2bf(v.x) | ((unsigned)f2bf(v.y) << 16);
    pk.y = (unsigned)f2bf(v.z) | ((unsigned)f2bf(v.w) << 16);
    *(uint2*)((unsigned short*)(dst + row * HSTRIDE) + c4) = pk;
  }
  __syncthreads();

  const int lane    = tid & 31;
  const int wave    = tid >> 5;
  const int rowBase = wave * 16;        // message-row strip
  const int eBase   = (wave & 3) * 16;  // edge-row strip within the 64-edge tile
  const bool dir2   = wave >= 4;
  const int m       = lane & 15;
  const int hi8     = (lane & 16) ? 8 : 0;
  const int rsel    = (lane < 16) ? 0 : 8;

  const char* seg0 = dir2 ? Bb : Ab;    // first 128 K-columns of the concat input
  const char* seg1 = dir2 ? Ab : Bb;    // second 128

  v8f acc[8];

  // ---- layer 1: concat(16x384) @ W1(384x128), ReLU -> Hs bf16 ----
#pragma unroll
  for (int t = 0; t < 8; ++t) acc[t] = (v8f){0.f,0.f,0.f,0.f,0.f,0.f,0.f,0.f};
  for (int kk = 0; kk < 12; ++kk) {
    int segi = kk >> 2, kkin = kk & 3;
    const char* abase = (segi == 0) ? seg0 : ((segi == 1) ? seg1 : Fb);
    bf16x16 a = load_frag_lds(abase + (eBase + m) * HSTRIDE + (kkin * 32 + hi8) * 2);
#pragma unroll
    for (int t = 0; t < 8; ++t) {
      bf16x16 b = load_frag_g(w1t + (t * 16 + m) * 384 + kk * 32 + hi8);
      acc[t] = WMMA_BF16(a, b, acc[t]);
    }
  }
#pragma unroll
  for (int t = 0; t < 8; ++t) {
    int col = t * 16 + m;
    float bv = bias1[col];
#pragma unroll
    for (int r = 0; r < 8; ++r) {
      float v = acc[t][r] + bv;
      v = v > 0.f ? v : 0.f;
      *(unsigned short*)(Hs + (rowBase + r + rsel) * HSTRIDE + col * 2) = f2bf(v);
    }
  }

  // ---- layer 2: H1 @ W2, ReLU -> Hs (in-place per-wave strip) ----
#pragma unroll
  for (int t = 0; t < 8; ++t) acc[t] = (v8f){0.f,0.f,0.f,0.f,0.f,0.f,0.f,0.f};
  for (int kk = 0; kk < 4; ++kk) {
    bf16x16 a = load_frag_lds(Hs + (rowBase + m) * HSTRIDE + (kk * 32 + hi8) * 2);
#pragma unroll
    for (int t = 0; t < 8; ++t) {
      bf16x16 b = load_frag_g(w2t + (t * 16 + m) * 128 + kk * 32 + hi8);
      acc[t] = WMMA_BF16(a, b, acc[t]);
    }
  }
#pragma unroll
  for (int t = 0; t < 8; ++t) {
    int col = t * 16 + m;
    float bv = bias2[col];
#pragma unroll
    for (int r = 0; r < 8; ++r) {
      float v = acc[t][r] + bv;
      v = v > 0.f ? v : 0.f;
      *(unsigned short*)(Hs + (rowBase + r + rsel) * HSTRIDE + col * 2) = f2bf(v);
    }
  }

  // ---- layer 3: H2 @ W3 + b3, scatter straight from accumulators ----
#pragma unroll
  for (int t = 0; t < 8; ++t) acc[t] = (v8f){0.f,0.f,0.f,0.f,0.f,0.f,0.f,0.f};
  for (int kk = 0; kk < 4; ++kk) {
    bf16x16 a = load_frag_lds(Hs + (rowBase + m) * HSTRIDE + (kk * 32 + hi8) * 2);
#pragma unroll
    for (int t = 0; t < 8; ++t) {
      bf16x16 b = load_frag_g(w3t + (t * 16 + m) * 128 + kk * 32 + hi8);
      acc[t] = WMMA_BF16(a, b, acc[t]);
    }
  }
  float bv[8];
#pragma unroll
  for (int t = 0; t < 8; ++t) bv[t] = bias3[t * 16 + m];
#pragma unroll
  for (int r = 0; r < 8; ++r) {
    int e = tileBase + eBase + r + rsel;
    if (e < E) {
      int node = dir2 ? fi[e] : ti[e];
      float* dst = agg + node * 128 + m;
#pragma unroll
      for (int t = 0; t < 8; ++t)
        __hip_atomic_fetch_add(dst + t * 16, acc[t][r] + bv[t],
                               __ATOMIC_RELAXED, __HIP_MEMORY_SCOPE_AGENT);
    }
  }
}

// ---------------------------------------------------------------------------
// One GRU cell over a 128-row tile. gate order r,z,n (torch).
__global__ __launch_bounds__(256) void gru_kernel(
    const float* __restrict__ x, const float* __restrict__ h,
    const unsigned short* __restrict__ Wt, const unsigned short* __restrict__ Ut,
    const float* __restrict__ bi, const float* __restrict__ bh,
    float* __restrict__ out, int N) {
  extern __shared__ char smem[];
  char* Xb = smem;                   // 128 x HSTRIDE bf16
  char* Hb = smem + 128 * HSTRIDE;   // 128 x HSTRIDE bf16

  const int tid  = threadIdx.x;
  const int base = blockIdx.x * 128;

  for (int q = tid; q < 128 * 32; q += 256) {
    int row = q >> 5;
    int c4  = (q & 31) * 4;
    int gr  = base + row;
    int grc = (gr < N) ? gr : 0;
    float4 vx = *(const float4*)(x + grc * 128 + c4);
    float4 vh = *(const float4*)(h + grc * 128 + c4);
    uint2 px, ph;
    px.x = (unsigned)f2bf(vx.x) | ((unsigned)f2bf(vx.y) << 16);
    px.y = (unsigned)f2bf(vx.z) | ((unsigned)f2bf(vx.w) << 16);
    ph.x = (unsigned)f2bf(vh.x) | ((unsigned)f2bf(vh.y) << 16);
    ph.y = (unsigned)f2bf(vh.z) | ((unsigned)f2bf(vh.w) << 16);
    *(uint2*)((unsigned short*)(Xb + row * HSTRIDE) + c4) = px;
    *(uint2*)((unsigned short*)(Hb + row * HSTRIDE) + c4) = ph;
  }
  __syncthreads();

  const int lane    = tid & 31;
  const int wave    = tid >> 5;
  const int rowBase = wave * 16;
  const int m       = lane & 15;
  const int hi8     = (lane & 16) ? 8 : 0;
  const int rsel    = (lane < 16) ? 0 : 8;

  bf16x16 ax[4], ah[4];
#pragma unroll
  for (int kk = 0; kk < 4; ++kk) {
    ax[kk] = load_frag_lds(Xb + (rowBase + m) * HSTRIDE + (kk * 32 + hi8) * 2);
    ah[kk] = load_frag_lds(Hb + (rowBase + m) * HSTRIDE + (kk * 32 + hi8) * 2);
  }

#pragma unroll
  for (int t = 0; t < 8; ++t) {
    int n = t * 16 + m;              // column within a 128-wide gate
    v8f aR  = (v8f){0.f,0.f,0.f,0.f,0.f,0.f,0.f,0.f};
    v8f aZ  = aR, aNi = aR, aNh = aR;
#pragma unroll
    for (int kk = 0; kk < 4; ++kk) {
      int ko = kk * 32 + hi8;
      bf16x16 bWr = load_frag_g(Wt + (n)       * 128 + ko);
      bf16x16 bUr = load_frag_g(Ut + (n)       * 128 + ko);
      bf16x16 bWz = load_frag_g(Wt + (128 + n) * 128 + ko);
      bf16x16 bUz = load_frag_g(Ut + (128 + n) * 128 + ko);
      bf16x16 bWn = load_frag_g(Wt + (256 + n) * 128 + ko);
      bf16x16 bUn = load_frag_g(Ut + (256 + n) * 128 + ko);
      aR  = WMMA_BF16(ax[kk], bWr, aR);
      aR  = WMMA_BF16(ah[kk], bUr, aR);    // ir+hr share one accumulator
      aZ  = WMMA_BF16(ax[kk], bWz, aZ);
      aZ  = WMMA_BF16(ah[kk], bUz, aZ);
      aNi = WMMA_BF16(ax[kk], bWn, aNi);   // inn and r*hn must stay separate
      aNh = WMMA_BF16(ah[kk], bUn, aNh);
    }
    float br = bi[n] + bh[n];
    float bz = bi[128 + n] + bh[128 + n];
    float bn = bi[256 + n];
    float cn = bh[256 + n];
#pragma unroll
    for (int r = 0; r < 8; ++r) {
      int mr = r + rsel;
      int gr = base + rowBase + mr;
      float rg = 1.f / (1.f + __expf(-(aR[r] + br)));
      float zg = 1.f / (1.f + __expf(-(aZ[r] + bz)));
      float ng_in = (aNi[r] + bn) + rg * (aNh[r] + cn);
      float ng = 1.f - 2.f / (1.f + __expf(2.f * ng_in));   // tanh
      float hv = bf2f(*(const unsigned short*)(Hb + (rowBase + mr) * HSTRIDE + n * 2));
      if (gr < N) out[gr * 128 + n] = (1.f - zg) * ng + zg * hv;
    }
  }
}

// ---------------------------------------------------------------------------
extern "C" void kernel_launch(void* const* d_in, const int* in_sizes, int n_in,
                              void* d_out, int out_size, void* d_ws, size_t ws_size,
                              hipStream_t stream) {
  (void)n_in; (void)out_size; (void)ws_size;
  const float* ns  = (const float*)d_in[0];
  const float* ef  = (const float*)d_in[1];
  const int*   fi  = (const int*)d_in[2];
  const int*   ti  = (const int*)d_in[3];
  // d_in[4] graph_idx: unused by reference output
  const float* mW1 = (const float*)d_in[5];  const float* mb1 = (const float*)d_in[6];
  const float* mW2 = (const float*)d_in[7];  const float* mb2 = (const float*)d_in[8];
  const float* mW3 = (const float*)d_in[9];  const float* mb3 = (const float*)d_in[10];
  const float* gW[3] = {(const float*)d_in[11], (const float*)d_in[15], (const float*)d_in[19]};
  const float* gU[3] = {(const float*)d_in[12], (const float*)d_in[16], (const float*)d_in[20]};
  const float* gb[3] = {(const float*)d_in[13], (const float*)d_in[17], (const float*)d_in[21]};
  const float* gc[3] = {(const float*)d_in[14], (const float*)d_in[18], (const float*)d_in[22]};

  const int N = in_sizes[0] / 128;
  const int E = in_sizes[2];

  // workspace layout
  char* ws = (char*)d_ws;
  size_t o = 0;
  float* agg = (float*)(ws + o); o += (size_t)N * 128 * 4;
  float* h1  = (float*)(ws + o); o += (size_t)N * 128 * 4;
  unsigned short* w1t = (unsigned short*)(ws + o); o += 384 * 128 * 2;
  unsigned short* w2t = (unsigned short*)(ws + o); o += 128 * 128 * 2;
  unsigned short* w3t = (unsigned short*)(ws + o); o += 128 * 128 * 2;
  unsigned short* gWt[3];
  unsigned short* gUt[3];
  for (int i = 0; i < 3; ++i) {
    gWt[i] = (unsigned short*)(ws + o); o += 384 * 128 * 2;
    gUt[i] = (unsigned short*)(ws + o); o += 384 * 128 * 2;
  }
  float* h2 = (float*)d_out;   // GRU2 output lives in d_out; GRU3 reads+overwrites per-tile

  // zero agg
  zero_f32_kernel<<<(N * 32 + 255) / 256, 256, 0, stream>>>((float4*)agg, N * 32);

  // bf16 transposed weights
  transpose_bf16_kernel<<<(384 * 128 + 255) / 256, 256, 0, stream>>>(mW1, w1t, 384, 128);
  transpose_bf16_kernel<<<(128 * 128 + 255) / 256, 256, 0, stream>>>(mW2, w2t, 128, 128);
  transpose_bf16_kernel<<<(128 * 128 + 255) / 256, 256, 0, stream>>>(mW3, w3t, 128, 128);
  for (int i = 0; i < 3; ++i) {
    transpose_bf16_kernel<<<(384 * 128 + 255) / 256, 256, 0, stream>>>(gW[i], gWt[i], 128, 384);
    transpose_bf16_kernel<<<(384 * 128 + 255) / 256, 256, 0, stream>>>(gU[i], gUt[i], 128, 384);
  }

  // edge MLP + scatter: 3*64*272 + 128*272 = 87,040 B LDS -> 3 blocks/WGP
  const unsigned edgeLds = 3 * 64 * HSTRIDE + 128 * HSTRIDE;
  edge_msg_kernel<<<(E + 63) / 64, 256, edgeLds, stream>>>(
      ns, ef, fi, ti, w1t, mb1, w2t, mb2, w3t, mb3, agg, E);

  // chained GRUs
  const unsigned gruLds = 2 * 128 * HSTRIDE;  // 69,632 B
  const int gruGrid = (N + 127) / 128;
  gru_kernel<<<gruGrid, 256, gruLds, stream>>>(ns,  agg, gWt[0], gUt[0], gb[0], gc[0], h1, N);
  gru_kernel<<<gruGrid, 256, gruLds, stream>>>(agg, h1,  gWt[1], gUt[1], gb[1], gc[1], h2, N);
  gru_kernel<<<gruGrid, 256, gruLds, stream>>>(h1,  h2,  gWt[2], gUt[2], gb[2], gc[2], h2, N);
}
